// CosineSimilarityWeightedAverage_23888608101003
// MI455X (gfx1250) — compile-verified, matrix operand order checked by reading, not run on hardware
//
#include <hip/hip_runtime.h>

typedef __attribute__((ext_vector_type(16))) _Float16 v16h;
typedef __attribute__((ext_vector_type(8)))  _Float16 v8h;
typedef __attribute__((ext_vector_type(8)))  float    v8f;

#define C_DIM 4096
#define D_DIM 64
#define B_DIM 4
#define NKB   (C_DIM / 64)
// exp(10*cos - 10) == exp2(cos*K - K), K = 10*log2(e)
#define EXP_K 14.42695040888963f

static __device__ __forceinline__ v16h cat16(v8h a, v8h b) {
  return __builtin_shufflevector(a, b, 0,1,2,3,4,5,6,7,8,9,10,11,12,13,14,15);
}

// one 16-byte async global->LDS copy per lane (GVS addressing), ASYNCcnt-tracked
static __device__ __forceinline__ void async_cp16(unsigned lds_off,
                                                  const void* sbase,
                                                  unsigned goff) {
  asm volatile("global_load_async_to_lds_b128 %0, %1, %2"
               :
               : "v"(lds_off), "v"(goff), "s"((unsigned long long)sbase)
               : "memory");
}

// ---------------- Kernel 1: row-normalize input -> f16 Qn [B*C, 64] ------------
__global__ void normalize_rows_k(const float* __restrict__ inp,
                                 _Float16* __restrict__ qn) {
  const int lane = threadIdx.x & 31;
  const int wave = threadIdx.x >> 5;
  const int row  = blockIdx.x * 8 + wave;
  const float* src = inp + (size_t)row * D_DIM;
  float a = src[lane];
  float b = src[lane + 32];
  float ss = a * a + b * b;
  #pragma unroll
  for (int m = 1; m < 32; m <<= 1) ss += __shfl_xor(ss, m);
  float rn = rsqrtf(ss);
  _Float16* dst = qn + (size_t)row * D_DIM;
  dst[lane]      = (_Float16)(a * rn);
  dst[lane + 32] = (_Float16)(b * rn);
}

// ------------- Kernel 2: weight/bias [C,D] f32 -> transposed f16 [D,C] ---------
__global__ void transpose_tables_k(const float* __restrict__ w,
                                   const float* __restrict__ bsrc,
                                   _Float16* __restrict__ wt,
                                   _Float16* __restrict__ bt) {
  const int i = blockIdx.x * blockDim.x + threadIdx.x;   // 0 .. C*D-1
  const int c = i & (C_DIM - 1);
  const int d = i >> 12;
  wt[(size_t)d * C_DIM + c] = (_Float16)w[(size_t)c * D_DIM + d];
  bt[(size_t)d * C_DIM + c] = (_Float16)bsrc[(size_t)c * D_DIM + d];
}

// ------------- Kernel 3: fused flash-style cosine attention -------------------
// 256 threads = 8 wave32s; 128 rows per block; quad-buffered async K/V tiles,
// 3 tiles in flight, one barrier per tile.
__launch_bounds__(256, 1)
__global__ void flash_cosine_k(const _Float16* __restrict__ qn,   // [B*C, 64]
                               const _Float16* __restrict__ wt,   // [64, C]
                               const _Float16* __restrict__ bt,   // [64, C]
                               const float* __restrict__ inp,     // [B*C, 64]
                               const float* __restrict__ wgl,     // [C, 64]
                               const float* __restrict__ bgl,     // [C, 64]
                               float* __restrict__ out) {         // [B*C, 64]
  // dynamic LDS: 4 buffer sets x (sK 8KB | sVw 8KB | sVb 8KB) = 98304 B
  extern __shared__ _Float16 dynbuf[];
  __shared__ _Float16 sP[8 * 16 * 64];   // per-wave P staging (16 KB)

  const int tid   = threadIdx.x;
  const int wave  = tid >> 5;
  const int lane  = tid & 31;
  const int col   = lane & 15;
  const int hi    = lane >> 4;
  const int batch = blockIdx.y;
  const int rowbase = blockIdx.x * 128 + wave * 16;

  const unsigned ldsBase = (unsigned)(size_t)(void*)dynbuf;  // addr[31:0] = LDS offset

  // Q tile in A-matrix layout
  const _Float16* qrow = qn + ((size_t)batch * C_DIM + rowbase + col) * D_DIM;
  v16h qa[2];
  #pragma unroll
  for (int ks = 0; ks < 2; ++ks) {
    v8h a0 = *(const v8h*)(qrow + ks * 32 + hi * 8);
    v8h a1 = *(const v8h*)(qrow + ks * 32 + 16 + hi * 8);
    qa[ks] = cat16(a0, a1);
  }

  v8f accw[4] = {};
  v8f accb[4] = {};
  float lpart[8];                       // per-lane partial row sums
  #pragma unroll
  for (int v = 0; v < 8; ++v) lpart[v] = 0.0f;

  _Float16* sPw = sP + wave * (16 * 64);

  // issue 6 async B128 copies per thread for tile kb into buffer set `set`
  auto issue_tile = [&](int kb, int set) {
    const _Float16* kbase = qn + ((size_t)batch * C_DIM + kb * 64) * D_DIM;
    const _Float16* wbase = wt + kb * 64;
    const _Float16* bbase = bt + kb * 64;
    const unsigned sk = ldsBase + (unsigned)set * 24576u;
    const unsigned sw = sk + 8192u;
    const unsigned sb = sk + 16384u;
    #pragma unroll
    for (int j = 0; j < 2; ++j) {
      unsigned u   = (unsigned)tid + (unsigned)j * 256u;   // 0..511
      unsigned d   = u >> 3;
      unsigned o   = (u & 7u) * 8u;
      unsigned l16 = u * 16u;
      async_cp16(sk + l16, kbase, l16);                       // contiguous 8 KB
      async_cp16(sw + l16, wbase, (d * (unsigned)C_DIM + o) * 2u);
      async_cp16(sb + l16, bbase, (d * (unsigned)C_DIM + o) * 2u);
    }
  };

  issue_tile(0, 0);
  issue_tile(1, 1);
  issue_tile(2, 2);

  for (int kb = 0; kb < NKB; ++kb) {
    // wait for tile kb's 6 copies (up to 2 later tiles may stay in flight)
    if      (kb + 2 < NKB) asm volatile("s_wait_asynccnt 0xc" ::: "memory");
    else if (kb + 1 < NKB) asm volatile("s_wait_asynccnt 0x6" ::: "memory");
    else                   asm volatile("s_wait_asynccnt 0x0" ::: "memory");
    // one barrier: tile kb visible to all waves AND all waves finished
    // reading the buffer that tile kb+3 will overwrite (read at kb-1)
    __syncthreads();
    if (kb + 3 < NKB) issue_tile(kb + 3, (kb + 3) & 3);

    const _Float16* sK  = dynbuf + (kb & 3) * 12288;       // [k][d]
    const _Float16* sVw = sK + 4096;                       // [d][k]
    const _Float16* sVb = sK + 8192;                       // [d][k]

    // ---- S = Q . K^T (raw cosines, f32) ----
    v8f st[4];
    #pragma unroll
    for (int nt = 0; nt < 4; ++nt) {
      v8f s = {};
      #pragma unroll
      for (int ks = 0; ks < 2; ++ks) {
        const _Float16* kr = sK + (nt * 16 + col) * 64 + ks * 32 + hi * 8;
        v8h b0 = *(const v8h*)kr;
        v8h b1 = *(const v8h*)(kr + 16);
        s = __builtin_amdgcn_wmma_f32_16x16x32_f16(
                false, qa[ks], false, cat16(b0, b1), (short)0, s, false, false);
      }
      st[nt] = s;
    }

    // ---- softmax numerator with FIXED max (cos <= 1 -> logit <= 10);
    //      per-lane partial sums only, cross-lane reduce after the loop ----
    #pragma unroll
    for (int v = 0; v < 8; ++v) {
      #pragma unroll
      for (int nt = 0; nt < 4; ++nt) {
        float p = __builtin_amdgcn_exp2f(fmaf(st[nt][v], EXP_K, -EXP_K));
        st[nt][v] = p;
        lpart[v] += p;
      }
      #pragma unroll
      for (int nt = 0; nt < 4; ++nt)
        sPw[(v + 8 * hi) * 64 + nt * 16 + col] = (_Float16)st[nt][v];
    }

    // ---- reload P in A-matrix layout (wave-private, in-wave DS ordering) ----
    v16h pa[2];
    #pragma unroll
    for (int ks = 0; ks < 2; ++ks) {
      const _Float16* pr = sPw + col * 64 + ks * 32 + hi * 8;
      v8h a0 = *(const v8h*)pr;
      v8h a1 = *(const v8h*)(pr + 16);
      pa[ks] = cat16(a0, a1);
    }

    // ---- acc += P @ V for both tables ----
    #pragma unroll
    for (int nt = 0; nt < 4; ++nt) {
      #pragma unroll
      for (int ks = 0; ks < 2; ++ks) {
        const _Float16* vr = sVw + (nt * 16 + col) * 64 + ks * 32 + hi * 8;
        v8h w0 = *(const v8h*)vr;
        v8h w1 = *(const v8h*)(vr + 16);
        accw[nt] = __builtin_amdgcn_wmma_f32_16x16x32_f16(
                       false, pa[ks], false, cat16(w0, w1), (short)0, accw[nt], false, false);
        const _Float16* br = sVb + (nt * 16 + col) * 64 + ks * 32 + hi * 8;
        v8h c0 = *(const v8h*)br;
        v8h c1 = *(const v8h*)(br + 16);
        accb[nt] = __builtin_amdgcn_wmma_f32_16x16x32_f16(
                       false, pa[ks], false, cat16(c0, c1), (short)0, accb[nt], false, false);
      }
    }
  }

  // ---- final cross-lane row-sum reduction (once), then epilogue ----
  #pragma unroll
  for (int v = 0; v < 8; ++v) {
    float ps = lpart[v];
    ps += __shfl_xor(ps, 1);
    ps += __shfl_xor(ps, 2);
    ps += __shfl_xor(ps, 4);
    ps += __shfl_xor(ps, 8);
    const float linv = 1.0f / ps;
    const int c = rowbase + v + 8 * hi;
    #pragma unroll
    for (int nt = 0; nt < 4; ++nt) {
      const int d = nt * 16 + col;
      const size_t t   = (size_t)c * D_DIM + d;
      const size_t bt_ = (size_t)batch * C_DIM * D_DIM + t;
      out[bt_] = accw[nt][v] * linv * wgl[t] * inp[bt_] +
                 accb[nt][v] * linv * bgl[t];
    }
  }
}

extern "C" void kernel_launch(void* const* d_in, const int* in_sizes, int n_in,
                              void* d_out, int out_size, void* d_ws, size_t ws_size,
                              hipStream_t stream) {
  const float* inp = (const float*)d_in[0];
  const float* w   = (const float*)d_in[1];
  const float* b   = (const float*)d_in[2];
  const float* wg  = (const float*)d_in[3];
  const float* bg  = (const float*)d_in[4];
  float* out = (float*)d_out;

  const size_t qn_bytes = (size_t)B_DIM * C_DIM * D_DIM * sizeof(_Float16); // 2 MB
  const size_t tb_bytes = (size_t)C_DIM * D_DIM * sizeof(_Float16);         // 512 KB
  _Float16* qn = (_Float16*)d_ws;
  _Float16* wt = (_Float16*)((char*)d_ws + qn_bytes);
  _Float16* bt = (_Float16*)((char*)d_ws + qn_bytes + tb_bytes);

  normalize_rows_k<<<B_DIM * C_DIM / 8, 256, 0, stream>>>(inp, qn);
  transpose_tables_k<<<C_DIM * D_DIM / 256, 256, 0, stream>>>(w, b, wt, bt);

  dim3 grid(C_DIM / 128, B_DIM);
  flash_cosine_k<<<grid, 256, 4 * 24576, stream>>>(qn, wt, bt, inp, wg, bg, out);
}
